// TriParallelMade_36936718746163
// MI455X (gfx1250) — matrix operand order repaired and checked by reading, not compiled, not run on hardware
//
#include <hip/hip_runtime.h>
#include <hip/hip_bf16.h>

// ---------------------------------------------------------------------------
// MADE (masked MLP) on folded 64x64 lattice, B=128 — bf16 WMMA streaming GEMMs
// with Tensor-Data-Mover (TDM) double-buffered activation staging.
//  * Weights fp32 in HBM, streamed once (NT loads), converted to bf16 in regs,
//    fed as the WMMA B operand (row-major [Nout,K] == B-operand layout).
//  * Activations bf16 end-to-end; A tiles DMA'd global->LDS by TDM
//    (tensor_load_to_lds, TENSORcnt, D# pad feature provides bank-spread
//    row padding), ping-pong overlapped with WMMA compute.
//  * Autoregressive mask == per-row K threshold on contiguous level groups:
//    whole K-chunks above the block's max threshold are skipped; thresholds
//    are 16-aligned except folded rows p<16 (exactly one wave -> exact path).
// ---------------------------------------------------------------------------

typedef __bf16 bf16_t;
typedef __attribute__((ext_vector_type(16))) __bf16 v16bf;
typedef __attribute__((ext_vector_type(8)))  float  v8f;
typedef __attribute__((ext_vector_type(4)))  float  f32x4;
typedef __attribute__((ext_vector_type(4)))  unsigned u32x4;
typedef __attribute__((ext_vector_type(8)))  unsigned u32x8;

#define LFOLD   64
#define NFOLD   4096      // L*L
#define BATCH   128
#define WIDTH2N 8192      // WIDTH * NFOLD
#define KCHUNK  64        // K elements staged per TDM tile
#define APAD    72        // LDS row stride in bf16 (64 + 8 pad = 144B)

// ---- folded-order helpers (quad-tree fold of the lattice) ------------------

__device__ __forceinline__ int site_to_pos(int site) {
    int r = site >> 6, c = site & (LFOLD - 1);
    int m = r | c;
    if (m == 0) return 0;
    int i = __ffs(m) - 1;            // ctz: sublattice level (s = 2^i)
    int j = 5 - i;                   // lvl-level (group base = 4^j)
    int bitr = (r >> i) & 1, bitc = (c >> i) & 1;
    int cls = (bitr & bitc) ? 0 : (bitr ? 1 : 2);
    int rank = ((r >> (i + 1)) << j) + (c >> (i + 1));
    return ((1 + cls) << (2 * j)) + rank;
}

__device__ __forceinline__ int pos_to_site(int p) {
    if (p == 0) return 0;
    int b = 31 - __clz(p);
    int j = b >> 1;
    int base = 1 << (2 * j);
    int seg  = (p >> (2 * j)) - 1;   // 0,1,2
    int rank = p & (base - 1);
    int i = 5 - j;
    int s = 1 << i;
    int r = ((rank >> j) << (i + 1)) + ((seg != 2) ? s : 0);
    int c = ((rank & ((1 << j) - 1)) << (i + 1)) + ((seg != 1) ? s : 0);
    return r * LFOLD + c;
}

// K-threshold of folded row p: exclusive -> group start, inclusive -> group end
__device__ __forceinline__ int row_threshold(int p, int inclusive) {
    if (p == 0) return inclusive ? 1 : 0;
    int b = 31 - __clz(p);
    int j2 = (b >> 1) << 1;                 // 2*j
    int start = (p >> j2) << j2;
    return inclusive ? start + (1 << j2) : start;
}

// next unskipped chunk start (mask allows (k mod 4096) < Tmax; Tmax >= 1)
__device__ __forceinline__ int next_k(int k, int K, int Tmax) {
    k += KCHUNK;
    if (k < K && (k & (NFOLD - 1)) >= Tmax)
        k = (k & ~(NFOLD - 1)) + NFOLD;     // jump to next 4096 boundary
    return k;
}

// ---- TDM: DMA a [128 x KCHUNK] bf16 tile (row stride K) into padded LDS ----
// D# per CDNA5 ISA: group0 = {count, lds_addr, global_addr, type},
// group1 = {data_size=2B, pad 32dw->4dw, dims/strides}. 2D tile: groups 2/3=0.

__device__ __forceinline__ void tdm_issue(const bf16_t* gsrc, unsigned lds_off,
                                          int K) {
    unsigned long long ga = (unsigned long long)(uintptr_t)gsrc;
    u32x4 g0;
    g0[0] = 1u;                                   // count=1, user descriptor
    g0[1] = lds_off;                              // lds_addr (bytes)
    g0[2] = (unsigned)ga;                         // global_addr[31:0]
    g0[3] = (unsigned)((ga >> 32) & 0x01FFFFFFu)  // global_addr[56:32]
          | (2u << 30);                           // type = 2 ("image")
    u32x8 g1;
    g1[0] = (1u << 16)                            // data_size = 2 bytes
          | (1u << 20)                            // pad_enable
          | (4u << 22)                            // pad_interval: 32 dwords
          | (3u << 25);                           // pad_amount : 4 dwords
    g1[1] = ((unsigned)K & 0xFFFFu) << 16;        // tensor_dim0[15:0]
    g1[2] = ((unsigned)K >> 16)                   // tensor_dim0[31:16]
          | (128u << 16);                         // tensor_dim1[15:0] = 128
    g1[3] = ((unsigned)KCHUNK << 16);             // tile_dim0 = 64
    g1[4] = 128u;                                 // tile_dim1 = 128, tile_dim2=0
    g1[5] = (unsigned)K;                          // tensor_dim0_stride[31:0]
    g1[6] = 0u;
    g1[7] = 0u;
    u32x4 gz = (u32x4)0u;
    asm volatile("tensor_load_to_lds %0, %1, %2, %3"
                 :: "s"(g0), "s"(g1), "s"(gz), "s"(gz)
                 : "memory");
}

// ---- fold: xf[b, pos(site)] = bf16(x[b, site]) -----------------------------

__global__ void made_fold(const float* __restrict__ x, bf16_t* __restrict__ xf) {
    int idx  = blockIdx.x * blockDim.x + threadIdx.x;   // b*4096 + site
    int b    = idx >> 12;
    int site = idx & (NFOLD - 1);
    xf[(b << 12) + site_to_pos(site)] = (bf16_t)x[idx];
}

// ---- masked GEMM: out[M=128, Nout] = act( A[128,K] @ (mask.W)^T + bias ) ---
// Block: 128 threads = 4 waves; wave computes a 128x16 stripe (8 C tiles).
// Grid.x = Nout/64.  INCLUSIVE: lvl<= vs lvl<.  ACT: 0=PReLU->bf16, 1=sigmoid
// + fused unfold -> f32 d_out.

template <int INCLUSIVE, int ACT>
__global__ __launch_bounds__(128)
void made_gemm(const bf16_t* __restrict__ A, const float* __restrict__ W,
               const float* __restrict__ bias, const float* __restrict__ alpha,
               void* __restrict__ outv, int K, int Nout) {
    __shared__ __align__(16) bf16_t sA[2][BATCH * APAD];   // 2 x 18432 B

    const int tid    = threadIdx.x;
    const int wave   = tid >> 5;
    const int lane   = tid & 31;
    const int lane16 = lane & 15;
    const int khalf  = lane >> 4;                 // 0: K 0..15, 1: K 16..31
    const int n_base = blockIdx.x * 64;

    // scalarized wave id: TDM ignores EXEC, so its guard must be a true
    // scalar branch, never an exec-mask.
    const int wv = __builtin_amdgcn_readfirstlane(wave);

    const int wr = n_base + wave * 16 + lane16;   // this lane's output neuron
    const int p  = wr & (NFOLD - 1);              // folded position of neuron
    const int T  = row_threshold(p, INCLUSIVE);   // allowed (k mod 4096) < T

    // wave-uniform: this wave owns folded rows p=0..15 (sub-16 thresholds)
    const int waveExact = __builtin_amdgcn_readfirstlane(
        (((n_base + wave * 16) & (NFOLD - 1)) == 0) ? 1 : 0);
    // block-uniform chunk skip bound (thresholds monotone over the block)
    const int Tmax = row_threshold((n_base + 63) & (NFOLD - 1), INCLUSIVE);

    const unsigned ldsoff0 = (unsigned)(uintptr_t)&sA[0][0];
    const unsigned ldsoff1 = (unsigned)(uintptr_t)&sA[1][0];

    v8f acc[8] = {};                              // 8 M-tiles, one N column/lane

    union AF { v16bf v; u32x4 q[2]; };

    int k0 = 0;                                   // chunk 0 always allowed
    if (wv == 0) tdm_issue(A + k0, ldsoff0, K);   // prime the pipeline
    int par = 0;

    while (k0 < K) {
        const int k1 = next_k(k0, K, Tmax);

        if (wv == 0) __builtin_amdgcn_s_wait_tensorcnt(0);   // chunk k0 landed
        __syncthreads();                                     // publish to WG
        if (wv == 0 && k1 < K)                               // overlap next DMA
            tdm_issue(A + k1, (par ? ldsoff0 : ldsoff1), K);

        // prefetch this lane's weight row two chunks ahead (streaming)
        if (k0 + 2 * KCHUNK < K)
            __builtin_prefetch(W + (size_t)wr * K + k0 + 2 * KCHUNK, 0, 1);

        const bf16_t* sbuf = &sA[par][0];
#pragma unroll
        for (int ks = 0; ks < 2; ++ks) {
            // B operand: lane = column wr, 16 consecutive K (fp32->bf16, mask)
            const int kk = k0 + ks * 32 + khalf * 16;
            const int jm = kk & (NFOLD - 1);
            const f32x4* wp = (const f32x4*)(W + (size_t)wr * K + kk);
            union { f32x4 v[4]; float f[16]; } wu;
            wu.v[0] = __builtin_nontemporal_load(wp + 0);
            wu.v[1] = __builtin_nontemporal_load(wp + 1);
            wu.v[2] = __builtin_nontemporal_load(wp + 2);
            wu.v[3] = __builtin_nontemporal_load(wp + 3);
            union { v16bf v; unsigned u[8]; } bu;
#pragma unroll
            for (int e = 0; e < 16; ++e) bu.v[e] = (bf16_t)wu.f[e];
            if (waveExact) {                       // rare: sub-16 threshold
#pragma unroll
                for (int e = 0; e < 16; ++e)
                    if (jm + e >= T) bu.v[e] = (bf16_t)0.0f;
            } else {                               // 16-aligned: all-or-nothing
                unsigned m32 = (jm < T) ? 0xFFFFFFFFu : 0u;
#pragma unroll
                for (int i = 0; i < 8; ++i) bu.u[i] &= m32;
            }

            // A operands: issue all 16 ds_load_b128 first, then fence the
            // scheduler so it cannot sink loads down to their WMMAs — keeps
            // the DS pipe 16 deep with incremental s_wait_dscnt instead of
            // a full drain per WMMA.
            const bf16_t* sb = sbuf + lane16 * APAD + ks * 32 + khalf * 16;
            AF af[8];
#pragma unroll
            for (int t = 0; t < 8; ++t) {
                const u32x4* ap = (const u32x4*)(sb + t * 16 * APAD);
                af[t].q[0] = ap[0];
                af[t].q[1] = ap[1];
            }
            __builtin_amdgcn_sched_barrier(0);     // loads stay above WMMAs
#pragma unroll
            for (int t = 0; t < 8; ++t)
                acc[t] = __builtin_amdgcn_wmma_f32_16x16x32_bf16(
                    false, af[t].v, false, bu.v, (short)0, acc[t], false, false);
        }
        par ^= 1;
        k0 = k1;
    }

    // ---- epilogue (C layout: lane<16 -> M=r, lane>=16 -> M=r+8, N=lane16)
    const float bv = bias[wr];
    if (ACT == 0) {
        bf16_t* out = (bf16_t*)outv;
        const float av = alpha[wr];
#pragma unroll
        for (int t = 0; t < 8; ++t)
#pragma unroll
            for (int r = 0; r < 8; ++r) {
                int m = t * 16 + r + khalf * 8;
                float v = acc[t][r] + bv;
                v = fmaxf(v, 0.0f) + av * fminf(v, 0.0f);   // PReLU
                out[(size_t)m * Nout + wr] = (bf16_t)v;     // bf16 activations
            }
    } else {
        float* out = (float*)outv;
        const int site = pos_to_site(p);
#pragma unroll
        for (int t = 0; t < 8; ++t)
#pragma unroll
            for (int r = 0; r < 8; ++r) {
                int m = t * 16 + r + khalf * 8;
                float v = acc[t][r] + bv;
                v = 1.0f / (1.0f + __expf(-v));             // sigmoid
                if (p == 0) v = 0.5f;                       // xhat mask/bias
                out[(size_t)m * NFOLD + site] = v;          // fused unfold
            }
    }
}

// ---------------------------------------------------------------------------

extern "C" void kernel_launch(void* const* d_in, const int* in_sizes, int n_in,
                              void* d_out, int out_size, void* d_ws, size_t ws_size,
                              hipStream_t stream) {
    const float* x  = (const float*)d_in[0];
    const float* W0 = (const float*)d_in[1];
    const float* b0 = (const float*)d_in[2];
    const float* a1 = (const float*)d_in[3];
    const float* W1 = (const float*)d_in[4];
    const float* b1 = (const float*)d_in[5];
    const float* a2 = (const float*)d_in[6];
    const float* W2 = (const float*)d_in[7];
    const float* b2 = (const float*)d_in[8];

    bf16_t* ws = (bf16_t*)d_ws;
    bf16_t* xf = ws;                             // [128, 4096] bf16
    bf16_t* h0 = xf + (size_t)BATCH * NFOLD;     // [128, 8192] bf16
    bf16_t* h1 = h0 + (size_t)BATCH * WIDTH2N;   // [128, 8192] bf16

    made_fold<<<(BATCH * NFOLD) / 256, 256, 0, stream>>>(x, xf);

    // layer 0: exclusive mask, PReLU(a1):  [128,4096] -> [128,8192] bf16
    made_gemm<0, 0><<<WIDTH2N / 64, 128, 0, stream>>>(xf, W0, b0, a1, h0,
                                                      NFOLD, WIDTH2N);
    // layer 1: inclusive mask, PReLU(a2):  [128,8192] -> [128,8192] bf16
    made_gemm<1, 0><<<WIDTH2N / 64, 128, 0, stream>>>(h0, W1, b1, a2, h1,
                                                      WIDTH2N, WIDTH2N);
    // layer 2: inclusive mask, sigmoid + unfold -> d_out [128,64,64] f32
    made_gemm<1, 1><<<NFOLD / 64, 128, 0, stream>>>(h1, W2, b2, nullptr,
                                                    (float*)d_out,
                                                    WIDTH2N, NFOLD);
}
